// TPUGEMMLinear_62551903699095
// MI455X (gfx1250) — 8-bit matrix operand order repaired and checked by reading, compiled, not run on hardware
//
#include <hip/hip_runtime.h>


#define MAT_N   4096
#define TILE_M  128
#define TILE_N  128
#define TILE_K  64

typedef __attribute__((ext_vector_type(8))) int v8i;

__global__ void init_ws(unsigned* ws) {
    if (threadIdx.x < 2) ws[threadIdx.x] = 0u;
}

__global__ void absmax_kernel(const float* __restrict__ x,
                              unsigned* __restrict__ out, int n4) {
    __shared__ float wmax[8];
    const int tid    = blockIdx.x * blockDim.x + threadIdx.x;
    const int stride = gridDim.x * blockDim.x;
    const float4* x4 = (const float4*)x;

    float m = 0.0f;
    for (int i = tid; i < n4; i += stride) {
        float4 v = x4[i];
        m = fmaxf(m, fmaxf(fmaxf(fabsf(v.x), fabsf(v.y)),
                           fmaxf(fabsf(v.z), fabsf(v.w))));
    }
    #pragma unroll
    for (int off = 16; off > 0; off >>= 1)
        m = fmaxf(m, __shfl_xor(m, off, 32));

    const int lane = threadIdx.x & 31;
    const int wid  = threadIdx.x >> 5;
    if (lane == 0) wmax[wid] = m;
    __syncthreads();
    if (wid == 0) {
        m = (lane < (int)(blockDim.x >> 5)) ? wmax[lane] : 0.0f;
        #pragma unroll
        for (int off = 4; off > 0; off >>= 1)
            m = fmaxf(m, __shfl_xor(m, off, 32));
        if (lane == 0) atomicMax(out, __float_as_uint(m));
    }
}

__device__ __forceinline__ signed char quant1(float v, float inv) {
    return (signed char)__float2int_rn(fminf(fmaxf(v * inv, -127.0f), 127.0f));
}

__global__ void quant_rm(const float* __restrict__ x,
                         signed char* __restrict__ q,
                         const unsigned* __restrict__ amax, int n4) {
    const float scale = fmaxf(__uint_as_float(*amax), 1e-8f) * (1.0f / 127.0f);
    const float inv   = 1.0f / scale;
    const int tid     = blockIdx.x * blockDim.x + threadIdx.x;
    const int stride  = gridDim.x * blockDim.x;
    const float4* x4  = (const float4*)x;
    char4* q4         = (char4*)q;
    for (int i = tid; i < n4; i += stride) {
        float4 v = x4[i];
        char4 c;
        c.x = quant1(v.x, inv); c.y = quant1(v.y, inv);
        c.z = quant1(v.z, inv); c.w = quant1(v.w, inv);
        q4[i] = c;
    }
}

__global__ void quantT(const float* __restrict__ w,
                       signed char* __restrict__ wt,
                       const unsigned* __restrict__ amax) {
    __shared__ signed char tile[32][33];
    const float inv = 127.0f / fmaxf(__uint_as_float(*amax), 1e-8f);
    const int n0 = blockIdx.x * 32;
    const int k0 = blockIdx.y * 32;
    #pragma unroll
    for (int i = 0; i < 32; i += 8) {
        const int k = k0 + threadIdx.y + i;
        const int n = n0 + threadIdx.x;
        tile[threadIdx.y + i][threadIdx.x] = quant1(w[(size_t)k * MAT_N + n], inv);
    }
    __syncthreads();
    #pragma unroll
    for (int i = 0; i < 32; i += 8) {
        const int n = n0 + threadIdx.y + i;
        const int k = k0 + threadIdx.x;
        wt[(size_t)n * MAT_N + k] = tile[threadIdx.x][threadIdx.y + i];
    }
}

__global__ __launch_bounds__(256)
void gemm_i8_wmma(const signed char* __restrict__ qx,
                  const signed char* __restrict__ qwt,
                  const unsigned*    __restrict__ amax2,
                  const float*       __restrict__ bias,
                  float*             __restrict__ out) {
    __shared__ __align__(16) signed char sA[2][TILE_M][TILE_K];
    __shared__ __align__(16) signed char sB[2][TILE_N][TILE_K];

    const int tid    = threadIdx.x;
    const int lane   = tid & 31;
    const int wid    = tid >> 5;
    const int wave_m = wid >> 1;
    const int wave_n = wid & 1;
    const int m0     = blockIdx.y * TILE_M;
    const int n0     = blockIdx.x * TILE_N;

    const int ld_row = tid >> 1;
    const int ld_col = (tid & 1) * 32;
    const unsigned a_goff = (unsigned)((m0 + ld_row) * MAT_N + ld_col);
    const unsigned b_goff = (unsigned)((n0 + ld_row) * MAT_N + ld_col);

    const int mrow = lane & 15;
    const int kb8  = (lane >> 4) * 8;
    const int kb16 = (lane >> 4) * 16;

    auto issue_tile = [&](int buf, int k0) {
        unsigned lA = (unsigned)(unsigned long long)&sA[buf][ld_row][ld_col];
        unsigned lB = (unsigned)(unsigned long long)&sB[buf][ld_row][ld_col];
        unsigned gA = a_goff + (unsigned)k0;
        unsigned gB = b_goff + (unsigned)k0;
        asm volatile(
            "global_load_async_to_lds_b128 %0, %1, %4 offset:0\n\t"
            "global_load_async_to_lds_b128 %0, %1, %4 offset:16\n\t"
            "global_load_async_to_lds_b128 %2, %3, %5 offset:0\n\t"
            "global_load_async_to_lds_b128 %2, %3, %5 offset:16"
            :
            : "v"(lA), "v"(gA), "v"(lB), "v"(gB), "s"(qx), "s"(qwt)
            : "memory");
    };

    const v8i vzero = {0, 0, 0, 0, 0, 0, 0, 0};
    v8i acc[2][4];
    #pragma unroll
    for (int mt = 0; mt < 2; ++mt)
        #pragma unroll
        for (int nt = 0; nt < 4; ++nt) acc[mt][nt] = vzero;

    auto compute_tile = [&](int cur) {
        v8i afrag[2];
        #pragma unroll
        for (int mt = 0; mt < 2; ++mt) {
            const signed char* p = &sA[cur][wave_m * 32 + mt * 16 + mrow][kb8];
            int2 w0 = *(const int2*)(p +  0);
            int2 w1 = *(const int2*)(p + 16);
            int2 w2 = *(const int2*)(p + 32);
            int2 w3 = *(const int2*)(p + 48);
            v8i a = {w0.x, w0.y, w1.x, w1.y, w2.x, w2.y, w3.x, w3.y};
            afrag[mt] = a;
        }
        #pragma unroll
        for (int nt = 0; nt < 4; ++nt) {
            const signed char* p = &sB[cur][wave_n * 64 + nt * 16 + mrow][kb8];
            int2 b0 = *(const int2*)(p +  0);
            int2 b1 = *(const int2*)(p + 16);
            int2 b2 = *(const int2*)(p + 32);
            int2 b3 = *(const int2*)(p + 48);
            v8i b = {b0.x, b0.y, b1.x, b1.y, b2.x, b2.y, b3.x, b3.y};
            acc[0][nt] = __builtin_amdgcn_wmma_i32_16x16x64_iu8(
                true, afrag[0], true, b, acc[0][nt], false, false);
            acc[1][nt] = __builtin_amdgcn_wmma_i32_16x16x64_iu8(
                true, afrag[1], true, b, acc[1][nt], false, false);
        }
    };

    issue_tile(0, 0);
    const int NK = MAT_N / TILE_K;

    for (int kt = 0; kt < NK - 1; ++kt) {
        const int cur = kt & 1;
        asm volatile("s_wait_asynccnt 0" ::: "memory");
        __syncthreads();
        issue_tile(cur ^ 1, (kt + 1) * TILE_K);
        compute_tile(cur);
    }
    asm volatile("s_wait_asynccnt 0" ::: "memory");
    __syncthreads();
    compute_tile((NK - 1) & 1);

    const float sx = fmaxf(__uint_as_float(amax2[0]), 1e-8f) * (1.0f / 127.0f);
    const float sw = fmaxf(__uint_as_float(amax2[1]), 1e-8f) * (1.0f / 127.0f);
    const float s  = sx * sw;

    const int colbase = lane & 15;
    const int rowsel  = (lane >> 4) * 8;
    #pragma unroll
    for (int nt = 0; nt < 4; ++nt) {
        const int col = n0 + wave_n * 64 + nt * 16 + colbase;
        const float bv = bias[col];
        #pragma unroll
        for (int mt = 0; mt < 2; ++mt) {
            const int rbase = m0 + wave_m * 32 + mt * 16 + rowsel;
            #pragma unroll
            for (int r = 0; r < 8; ++r)
                out[(size_t)(rbase + r) * MAT_N + col] =
                    (float)acc[mt][nt][r] * s + bv;
        }
    }
}

extern "C" void kernel_launch(void* const* d_in, const int* in_sizes, int n_in,
                              void* d_out, int out_size, void* d_ws, size_t ws_size,
                              hipStream_t stream) {
    const float* inputs = (const float*)d_in[0];
    const float* kern   = (const float*)d_in[1];
    const float* bias   = (const float*)d_in[2];
    float* out          = (float*)d_out;

    unsigned* ws_u   = (unsigned*)d_ws;
    signed char* qx  = (signed char*)d_ws + 256;
    signed char* qwt = qx + (size_t)MAT_N * MAT_N;

    const int n4 = (MAT_N * MAT_N) / 4;

    init_ws<<<1, 32, 0, stream>>>(ws_u);
    absmax_kernel<<<1024, 256, 0, stream>>>(inputs, ws_u + 0, n4);
    absmax_kernel<<<1024, 256, 0, stream>>>(kern,   ws_u + 1, n4);
    quant_rm<<<2048, 256, 0, stream>>>(inputs, qx, ws_u + 0, n4);
    quantT<<<dim3(MAT_N / 32, MAT_N / 32), dim3(32, 8), 0, stream>>>(kern, qwt, ws_u + 1);
    gemm_i8_wmma<<<dim3(MAT_N / TILE_N, MAT_N / TILE_M), 256, 0, stream>>>(
        qx, qwt, ws_u, bias, out);
}
